// EMAQuantizer_33389075759593
// MI455X (gfx1250) — matrix operand order repaired and checked by reading, not compiled, not run on hardware
//
#include <hip/hip_runtime.h>
#include <hip/hip_bf16.h>

#define DIM 256
#define KCODES 1024
#define NROWS 65536
#define TILE_ELEMS 4096    // 256 d * 16 codes (bf16) = 8 KB
#define CSTEP 32           // codes per iteration (two 16-wide column tiles)
#define NITER 32           // KCODES / CSTEP
#define STAGE_ELEMS 8192   // 2 tiles = 16 KB

typedef __attribute__((ext_vector_type(16))) __bf16 v16bf;
typedef __attribute__((ext_vector_type(8)))  float  v8f;

__device__ __forceinline__ __bf16 f2bf(float f) {
    union { float f; unsigned u; } v; v.f = f;
    unsigned r = v.u + 0x7FFFu + ((v.u >> 16) & 1u);   // round-to-nearest-even
    unsigned short h = (unsigned short)(r >> 16);
    return __builtin_bit_cast(__bf16, h);
}

// gfx1250 async global->LDS copy (16B), tracked by ASYNCcnt.
// VDST VGPR holds LDS byte offset (wave-relative); VADDR pair holds global addr (GV mode).
__device__ __forceinline__ void async_cp16(unsigned lds_off, unsigned long long gaddr) {
    asm volatile("global_load_async_to_lds_b128 %0, %1, off"
                 :: "v"(lds_off), "v"(gaddr) : "memory");
}
__device__ __forceinline__ void wait_async0() {
    asm volatile("s_wait_asynccnt 0x0" ::: "memory");
}

// --- Pass 1: tiled embT[tile][c%16][d] = bf16(emb[c][d]); enorm[c] = ||e_c||^2 ---
__global__ __launch_bounds__(256) void vq_prep(const float* __restrict__ emb,
                                               __bf16* __restrict__ embT,
                                               float* __restrict__ enorm) {
    int c = blockIdx.x;      // 0..1023 codes
    int d = threadIdx.x;     // 0..255  dims
    float e = emb[(size_t)c * DIM + d];
    embT[((size_t)(c >> 4) * 16 + (c & 15)) * DIM + d] = f2bf(e);
    __shared__ float sm[256];
    sm[d] = e * e; __syncthreads();
    for (int s = 128; s > 0; s >>= 1) { if (d < s) sm[d] += sm[d + s]; __syncthreads(); }
    if (d == 0) enorm[c] = sm[0];
}

// --- Pass 2: WMMA distance scores + argmin + gather + scatter-sums ---
__global__ __launch_bounds__(128) void vq_main(const float* __restrict__ x,
                                               const float* __restrict__ emb,
                                               const __bf16* __restrict__ embT,
                                               const float* __restrict__ enorm,
                                               float* __restrict__ out_q,
                                               float* __restrict__ out_idx,
                                               float* __restrict__ counts,
                                               float* __restrict__ esums,
                                               float* __restrict__ loss_acc) {
    __shared__ __align__(64) __bf16 bs[2][STAGE_ELEMS];  // double-buffered 16 KB code stages

    const int tid  = threadIdx.x;
    const int lane = tid & 31;
    const int wave = tid >> 5;
    const int row_base = (blockIdx.x * 4 + wave) * 16;   // 16 rows per wave, 64 per block

    // ---- Load A tiles (16x32 bf16 per D-chunk) into registers for all of D=256.
    // Layout: lanes 0-15 -> M=lane, intra-lane K = {b..b+15}, b=0;
    //         lanes 16-31 -> M=lane-16, b=16.
    const int half = lane >> 4;
    const int mrow = lane & 15;
    const float* xrow = x + (size_t)(row_base + mrow) * DIM;
    v16bf a[8];
#pragma unroll
    for (int dc = 0; dc < 8; ++dc) {
        int b0 = dc * 32 + 16 * half;
        float4 p0 = *(const float4*)(xrow + b0);
        float4 p1 = *(const float4*)(xrow + b0 + 4);
        float4 p2 = *(const float4*)(xrow + b0 + 8);
        float4 p3 = *(const float4*)(xrow + b0 + 12);
        a[dc][0]=f2bf(p0.x); a[dc][1]=f2bf(p0.y); a[dc][2]=f2bf(p0.z); a[dc][3]=f2bf(p0.w);
        a[dc][4]=f2bf(p1.x); a[dc][5]=f2bf(p1.y); a[dc][6]=f2bf(p1.z); a[dc][7]=f2bf(p1.w);
        a[dc][8]=f2bf(p2.x); a[dc][9]=f2bf(p2.y); a[dc][10]=f2bf(p2.z); a[dc][11]=f2bf(p2.w);
        a[dc][12]=f2bf(p3.x); a[dc][13]=f2bf(p3.y); a[dc][14]=f2bf(p3.z); a[dc][15]=f2bf(p3.w);
    }

    const unsigned lbase0 = (unsigned)(unsigned long long)&bs[0][0];
    const unsigned lbase1 = (unsigned)(unsigned long long)&bs[1][0];

    // Prime buffer 0 with code tiles {0,1} (16 KB, 128 B/thread, 8x16B async copies).
    {
        unsigned long long g = (unsigned long long)((const char*)embT + tid * 128);
        unsigned l = lbase0 + tid * 128;
#pragma unroll
        for (int i = 0; i < 8; ++i) async_cp16(l + 16 * i, g + 16 * i);
    }
    wait_async0();
    __syncthreads();

    float minv0[8]; int mini0[8];
    float minv1[8]; int mini1[8];
#pragma unroll
    for (int r = 0; r < 8; ++r) {
        minv0[r] = 3.4e38f; mini0[r] = 0;
        minv1[r] = 3.4e38f; mini1[r] = 0;
    }

    // ---- 32 iterations x 32 codes; two independent WMMA accumulator chains per iter;
    //      async-prefetch the next 16 KB stage into the other LDS buffer.
    for (int it = 0; it < NITER; ++it) {
        const int cur = it & 1;
        if (it + 1 < NITER) {
            unsigned long long g = (unsigned long long)
                ((const char*)embT + (size_t)(it + 1) * (STAGE_ELEMS * 2) + tid * 128);
            unsigned l = (cur ? lbase0 : lbase1) + tid * 128;
#pragma unroll
            for (int i = 0; i < 8; ++i) async_cp16(l + 16 * i, g + 16 * i);
        }

        const int c0 = it * CSTEP;
        const float en0 = enorm[c0 + (lane & 15)];
        const float en1 = enorm[c0 + 16 + (lane & 15)];
        v8f acc0 = {}, acc1 = {};
#pragma unroll
        for (int dc = 0; dc < 8; ++dc) {
            // B tiles 32x16: lane holds code n = lane&15; 16 contiguous d = dc*32+16*half.. in LDS.
            v16bf b0 = *(const v16bf*)&bs[cur][(lane & 15) * DIM + dc * 32 + 16 * half];
            v16bf b1 = *(const v16bf*)&bs[cur][TILE_ELEMS + (lane & 15) * DIM + dc * 32 + 16 * half];
            acc0 = __builtin_amdgcn_wmma_f32_16x16x32_bf16(
                    false, a[dc], false, b0, (short)0, acc0, false, false);
            acc1 = __builtin_amdgcn_wmma_f32_16x16x32_bf16(
                    false, a[dc], false, b1, (short)0, acc1, false, false);
        }
        // dist proxy = ||e||^2 - 2*score (||f||^2 is row-constant, irrelevant to argmin)
#pragma unroll
        for (int r = 0; r < 8; ++r) {
            float dv0 = __builtin_fmaf(-2.0f, acc0[r], en0);
            float dv1 = __builtin_fmaf(-2.0f, acc1[r], en1);
            if (dv0 < minv0[r]) { minv0[r] = dv0; mini0[r] = c0 + (lane & 15); }
            if (dv1 < minv1[r]) { minv1[r] = dv1; mini1[r] = c0 + 16 + (lane & 15); }
        }

        if (it + 1 < NITER) wait_async0();
        __syncthreads();
    }

    // ---- Merge the two column-tile streams (tie -> smaller index, i.e. stream 0).
    float minv[8]; int mini[8];
#pragma unroll
    for (int r = 0; r < 8; ++r) {
        bool t = (minv1[r] < minv0[r]);
        minv[r] = t ? minv1[r] : minv0[r];
        mini[r] = t ? mini1[r] : mini0[r];
    }

    // ---- Cross-lane argmin within each 16-lane half (C layout: lane = column/code).
#pragma unroll
    for (int m = 1; m <= 8; m <<= 1) {
#pragma unroll
        for (int r = 0; r < 8; ++r) {
            float ov = __shfl_xor(minv[r], m, 32);
            int   oi = __shfl_xor(mini[r], m, 32);
            if (ov < minv[r] || (ov == minv[r] && oi < mini[r])) { minv[r] = ov; mini[r] = oi; }
        }
    }

    // ---- Per-row: exact f32 gather of winner, loss partial, segment sums.
    float lloss = 0.0f;
#pragma unroll
    for (int rr = 0; rr < 16; ++rr) {
        int src = (rr < 8) ? 0 : 16;            // rows 0-7 in lanes 0-15, rows 8-15 in 16-31
        int idx = __shfl(mini[rr & 7], src, 32);
        int row = row_base + rr;
        const float* er = emb + (size_t)idx * DIM;
        const float* xr = x + (size_t)row * DIM;
        float* qr = out_q + (size_t)row * DIM;
        float* sr = esums + (size_t)idx * DIM;
#pragma unroll
        for (int d4 = lane * 4; d4 < DIM; d4 += 128) {
            float4 e = *(const float4*)(er + d4);
            float4 xv = *(const float4*)(xr + d4);
            *(float4*)(qr + d4) = e;
            float dx = e.x - xv.x, dy = e.y - xv.y, dz = e.z - xv.z, dw = e.w - xv.w;
            lloss += dx * dx + dy * dy + dz * dz + dw * dw;
            atomicAdd(sr + d4 + 0, xv.x);
            atomicAdd(sr + d4 + 1, xv.y);
            atomicAdd(sr + d4 + 2, xv.z);
            atomicAdd(sr + d4 + 3, xv.w);
        }
        if (lane == 0) {
            out_idx[row] = (float)idx;
            atomicAdd(&counts[idx], 1.0f);
        }
    }
#pragma unroll
    for (int m = 1; m < 32; m <<= 1) lloss += __shfl_xor(lloss, m, 32);
    if (lane == 0) atomicAdd(loss_acc, lloss);
}

// --- Pass 3: EMA update + normalization + loss scalar ---
__global__ __launch_bounds__(1024) void vq_final(const float* __restrict__ ema_cs,
                                                 const float* __restrict__ ema_emb,
                                                 const float* __restrict__ counts,
                                                 const float* __restrict__ esums,
                                                 const float* __restrict__ loss_acc,
                                                 float* __restrict__ out_necs,
                                                 float* __restrict__ out_nee,
                                                 float* __restrict__ out_nemb,
                                                 float* __restrict__ out_loss) {
    int k = threadIdx.x;   // 0..1023
    float necs = ema_cs[k] * 0.99f + 0.01f * counts[k];
    out_necs[k] = necs;
    __shared__ float sm[1024];
    sm[k] = necs; __syncthreads();
    for (int s = 512; s > 0; s >>= 1) { if (k < s) sm[k] += sm[k + s]; __syncthreads(); }
    float n = sm[0];
    float csz = (necs + 1e-5f) / (n + (float)KCODES * 1e-5f) * n;
    for (int d = 0; d < DIM; ++d) {
        float nee = ema_emb[(size_t)k * DIM + d] * 0.99f + 0.01f * esums[(size_t)k * DIM + d];
        out_nee[(size_t)k * DIM + d] = nee;
        out_nemb[(size_t)k * DIM + d] = nee / csz;
    }
    if (k == 0) out_loss[0] = 2.0f * loss_acc[0] / (float)((size_t)NROWS * DIM);
}

extern "C" void kernel_launch(void* const* d_in, const int* in_sizes, int n_in,
                              void* d_out, int out_size, void* d_ws, size_t ws_size,
                              hipStream_t stream) {
    const float* x       = (const float*)d_in[0];
    const float* emb     = (const float*)d_in[1];
    const float* ema_cs  = (const float*)d_in[2];
    const float* ema_emb = (const float*)d_in[3];

    // out layout: q (NROWS*DIM), idx (NROWS), loss (1), necs (K), nee (K*DIM), nemb (K*DIM)
    float* out      = (float*)d_out;
    float* out_q    = out;
    float* out_idx  = out + (size_t)NROWS * DIM;
    float* out_loss = out_idx + NROWS;
    float* out_necs = out_loss + 1;
    float* out_nee  = out_necs + KCODES;
    float* out_nemb = out_nee + (size_t)KCODES * DIM;

    char* w = (char*)d_ws;
    __bf16* embT  = (__bf16*)w;                                  // 512 KB (tiled)
    float* enorm  = (float*)(w + (size_t)KCODES * DIM * 2);      // 4 KB
    float* counts = enorm + KCODES;                              // 4 KB
    float* esums  = counts + KCODES;                             // 1 MB
    float* lossa  = esums + (size_t)KCODES * DIM;                // 4 B

    // zero the accumulators (counts, esums, loss are contiguous)
    hipMemsetAsync(counts, 0, (size_t)(KCODES + KCODES * DIM + 1) * sizeof(float), stream);

    vq_prep<<<KCODES, DIM, 0, stream>>>(emb, embT, enorm);
    vq_main<<<NROWS / 64, 128, 0, stream>>>(x, emb, embT, enorm,
                                            out_q, out_idx, counts, esums, lossa);
    vq_final<<<1, KCODES, 0, stream>>>(ema_cs, ema_emb, counts, esums, lossa,
                                       out_necs, out_nee, out_nemb, out_loss);
}
